// Model_21449066676862
// MI455X (gfx1250) — matrix operand order repaired and checked by reading, not compiled, not run on hardware
//
#include <hip/hip_runtime.h>
#include <hip/hip_bf16.h>

typedef __attribute__((ext_vector_type(16))) _Float16 v16h;
typedef __attribute__((ext_vector_type(8)))  float    v8f;

#define N_NODES 256
#define DIN     15
#define H       64
#define DOUT    10
#define NE      8192
#define EDIM    8
#define ETOT    (NE + N_NODES)   /* 8448 edges incl. self loops */
#define NH      (N_NODES * H)    /* 16384 */

// ---------- float <-> monotone uint key (for atomic max over floats) ----------
__device__ __forceinline__ unsigned fkey(float f) {
  unsigned u = __float_as_uint(f);
  return (u & 0x80000000u) ? ~u : (u | 0x80000000u);
}
__device__ __forceinline__ float funkey(unsigned k) {
  return (k & 0x80000000u) ? __uint_as_float(k ^ 0x80000000u) : __uint_as_float(~k);
}
#define NEG_INF_KEY 0x007FFFFFu  /* fkey(-inf) */

// ---------- init accumulators (graph-replay safe) ----------
__global__ void k_init(float* hbuf, float* v1, float* den, unsigned* mxk, float* v2) {
  int t = blockIdx.x * blockDim.x + threadIdx.x;
  if (t < NH)                           hbuf[t] = 0.f;
  else if (t < 2*NH)                    v1[t - NH] = 0.f;
  else if (t < 2*NH + N_NODES)          den[t - 2*NH] = 0.f;
  else if (t < 2*NH + 2*N_NODES)        mxk[t - 2*NH - N_NODES] = NEG_INF_KEY;
  else if (t < 2*NH + 2*N_NODES + H)    v2[t - (2*NH + 2*N_NODES)] = 0.f;
}

// ---------- y = relu(x @ W0 + b0)  [256,15]x[15,64] ----------
__global__ void k_lin0(const float* __restrict__ x, const float* __restrict__ W0,
                       const float* __restrict__ b0, float* __restrict__ y) {
  int t = blockIdx.x * blockDim.x + threadIdx.x;
  if (t >= NH) return;
  int n = t >> 6, hh = t & 63;
  float acc = b0[hh];
#pragma unroll
  for (int d = 0; d < DIN; ++d) acc += x[n * DIN + d] * W0[d * H + hh];
  y[t] = acc > 0.f ? acc : 0.f;
}

// ---------- edge_attr mean over E (for self-loop fill) ----------
__global__ void k_eamean(const float* __restrict__ ea, float* __restrict__ eamean) {
  __shared__ float s[256];
  int t = threadIdx.x;
  int d = t & 7, g = t >> 3;              // 32 partial sums per dim
  float acc = 0.f;
  for (int i = g; i < NE; i += 32) acc += ea[i * EDIM + d];
  s[t] = acc;
  __syncthreads();
  if (g == 0) {
    float sum = 0.f;
    for (int k = 0; k < 32; ++k) sum += s[k * 8 + d];
    eamean[d] = sum / (float)NE;
  }
}

// ---------- xl = y@Wl, xr = y@Wr via v_wmma_f32_16x16x32_f16 ----------
// one wave per 16x16 output tile; A fragment shared between both B matrices.
__global__ void __launch_bounds__(32) k_xlr(const float* __restrict__ y,
                                            const float* __restrict__ Wl,
                                            const float* __restrict__ Wr,
                                            float* __restrict__ xl,
                                            float* __restrict__ xr) {
  int lane = threadIdx.x;
  int mi = blockIdx.x >> 2;      // 16 row tiles
  int ni = blockIdx.x & 3;       // 4 col tiles
  int hi  = (lane >> 4) & 1;
  int lml = lane & 15;
  v8f cl = {}; v8f cr = {};
#pragma unroll
  for (int kb = 0; kb < H; kb += 32) {
    v16h a, bl, br;
#pragma unroll
    for (int v = 0; v < 8; ++v) {
      // A 16x32 f16 layout: lanes 0-15 M=0..15; v0..3 -> K 0..7, v4..7 -> K 16..23; +8 for hi lanes
      int kA = (v < 4 ? 2 * v : 16 + 2 * (v - 4)) + hi * 8;
      // B 32x16 f16 layout: same K order as A (v0..3 -> K 0..7, v4..7 -> K 16..23; +8 for hi lanes), N = lane&15
      int kB = (v < 4 ? 2 * v : 16 + 2 * (v - 4)) + hi * 8;
#pragma unroll
      for (int j = 0; j < 2; ++j) {
        a[2 * v + j]  = (_Float16)y[(mi * 16 + lml) * H + kb + kA + j];
        bl[2 * v + j] = (_Float16)Wl[(kb + kB + j) * H + ni * 16 + lml];
        br[2 * v + j] = (_Float16)Wr[(kb + kB + j) * H + ni * 16 + lml];
      }
    }
    cl = __builtin_amdgcn_wmma_f32_16x16x32_f16(false, a, false, bl, (short)0, cl, false, false);
    cr = __builtin_amdgcn_wmma_f32_16x16x32_f16(false, a, false, br, (short)0, cr, false, false);
  }
#pragma unroll
  for (int r = 0; r < 8; ++r) {   // C/D: VGPR r -> row r (+8 for hi lanes), col = lane&15
    int row = mi * 16 + hi * 8 + r;
    int col = ni * 16 + lml;
    xl[row * H + col] = cl[r];
    xr[row * H + col] = cr[r];
  }
}

// ---------- ef = ea_full @ We   [8448,8]x[8,64] ----------
__global__ void k_ef(const float* __restrict__ ea, const float* __restrict__ eamean,
                     const float* __restrict__ We, float* __restrict__ ef) {
  long t = (long)blockIdx.x * blockDim.x + threadIdx.x;
  if (t >= (long)ETOT * H) return;
  int e = (int)(t >> 6), hh = (int)(t & 63);
  float acc = 0.f;
#pragma unroll
  for (int d = 0; d < EDIM; ++d) {
    float av = (e < NE) ? ea[e * EDIM + d] : eamean[d];
    acc += av * We[d * H + hh];
  }
  ef[t] = acc;
}

// ---------- per-edge score + segment max ----------
__global__ void k_score(const int* __restrict__ ei, const float* __restrict__ xl,
                        const float* __restrict__ xr, const float* __restrict__ ef,
                        const float* __restrict__ att, float* __restrict__ sc,
                        unsigned* __restrict__ mxk) {
  int e = blockIdx.x * blockDim.x + threadIdx.x;
  if (e >= ETOT) return;
  int src = (e < NE) ? ei[e]      : (e - NE);
  int dst = (e < NE) ? ei[NE + e] : (e - NE);
  float s = 0.f;
  for (int hh = 0; hh < H; ++hh) {
    float m = xl[src * H + hh] + xr[dst * H + hh] + ef[e * H + hh];
    m = m > 0.f ? m : 0.2f * m;           // leaky relu
    s += m * att[hh];
  }
  sc[e] = s;
  atomicMax(&mxk[dst], fkey(s));
}

// ---------- ex = exp(score - max[dst]); den[dst] += ex ----------
__global__ void k_exden(const int* __restrict__ ei, const unsigned* __restrict__ mxk,
                        float* __restrict__ sc, float* __restrict__ den) {
  int e = blockIdx.x * blockDim.x + threadIdx.x;
  if (e >= ETOT) return;
  int dst = (e < NE) ? ei[NE + e] : (e - NE);
  float ex = expf(sc[e] - funkey(mxk[dst]));
  sc[e] = ex;
  atomicAdd(&den[dst], ex);
}

// ---------- h[dst] += (ex/den[dst]) * xl[src]  (per (edge, channel)) ----------
__global__ void k_aggr(const int* __restrict__ ei, const float* __restrict__ sc,
                       const float* __restrict__ den, const float* __restrict__ xl,
                       float* __restrict__ hbuf) {
  long t = (long)blockIdx.x * blockDim.x + threadIdx.x;
  if (t >= (long)ETOT * H) return;
  int e = (int)(t >> 6), hh = (int)(t & 63);
  int src = (e < NE) ? ei[e]      : (e - NE);
  int dst = (e < NE) ? ei[NE + e] : (e - NE);
  float alpha = sc[e] / den[dst];
  atomicAdd(&hbuf[dst * H + hh], alpha * xl[src * H + hh]);
}

// ---------- h += bg ----------
__global__ void k_hbias(float* __restrict__ hbuf, const float* __restrict__ bg) {
  int t = blockIdx.x * blockDim.x + threadIdx.x;
  if (t < NH) hbuf[t] += bg[t & 63];
}

// ---------- THE hot kernel: v1 = h_flat @ W1  (16384 x 16384 fp32, 1.07 GB) ----------
// grid (64 col-stripes, 16 row-chunks); each WG streams 1024 rows x 256 cols coalesced.
__global__ void __launch_bounds__(256) k_matvec1(const float* __restrict__ hb,
                                                 const float* __restrict__ W1,
                                                 float* __restrict__ v1) {
  __shared__ float vs[1024];
  int t = threadIdx.x;
  int j = blockIdx.x * 256 + t;                 // output column
  int ibase = blockIdx.y * 1024;                // input row chunk
#pragma unroll
  for (int q = 0; q < 4; ++q) vs[t + 256 * q] = hb[ibase + t + 256 * q];
  __syncthreads();
  float acc = 0.f;
  const float* p = W1 + (size_t)ibase * NH + j;
  for (int i = 0; i < 1024; ++i) {
    if ((i & 15) == 0) __builtin_prefetch(p + (size_t)16 * NH, 0, 0);  // global_prefetch_b8
    acc += vs[i] * (*p);
    p += NH;
  }
  atomicAdd(&v1[j], acc);
}

// ---------- v = relu(v + b) ----------
__global__ void k_reluB(float* __restrict__ v, const float* __restrict__ b, int n) {
  int t = blockIdx.x * blockDim.x + threadIdx.x;
  if (t < n) { float a = v[t] + b[t]; v[t] = a > 0.f ? a : 0.f; }
}

// ---------- v2 = v1 @ W2  (16384 x 64) ----------
__global__ void __launch_bounds__(256) k_matvec2(const float* __restrict__ v1,
                                                 const float* __restrict__ W2,
                                                 float* __restrict__ v2) {
  __shared__ float s[256];
  int t = threadIdx.x;
  int j = t & 63, g = t >> 6;                   // 4 i-groups per column
  int i0 = blockIdx.x * 512;
  float acc = 0.f;
  for (int i = i0 + g; i < i0 + 512; i += 4) acc += v1[i] * W2[i * H + j];
  s[t] = acc;
  __syncthreads();
  if (g == 0) atomicAdd(&v2[j], s[j] + s[64 + j] + s[128 + j] + s[192 + j]);
}

// ---------- out = v2 @ W3 + b3 ----------
__global__ void k_head(const float* __restrict__ v2, const float* __restrict__ W3,
                       const float* __restrict__ b3, float* __restrict__ out) {
  int d = threadIdx.x;
  if (d < DOUT) {
    float acc = b3[d];
    for (int jj = 0; jj < H; ++jj) acc += v2[jj] * W3[jj * DOUT + d];
    out[d] = acc;
  }
}

extern "C" void kernel_launch(void* const* d_in, const int* in_sizes, int n_in,
                              void* d_out, int out_size, void* d_ws, size_t ws_size,
                              hipStream_t stream) {
  (void)in_sizes; (void)n_in; (void)out_size; (void)ws_size;
  const float* x   = (const float*)d_in[0];
  const int*   ei  = (const int*)  d_in[1];
  const float* ea  = (const float*)d_in[2];
  const float* W0  = (const float*)d_in[3];
  const float* b0  = (const float*)d_in[4];
  const float* Wl  = (const float*)d_in[5];
  const float* Wr  = (const float*)d_in[6];
  const float* We  = (const float*)d_in[7];
  const float* att = (const float*)d_in[8];
  const float* bg  = (const float*)d_in[9];
  const float* W1  = (const float*)d_in[10];
  const float* b1  = (const float*)d_in[11];
  const float* W2  = (const float*)d_in[12];
  const float* b2  = (const float*)d_in[13];
  const float* W3  = (const float*)d_in[14];
  const float* b3  = (const float*)d_in[15];
  float* out = (float*)d_out;

  // scratch layout (floats) — ~2.5 MB total
  float*    ws     = (float*)d_ws;
  float*    y      = ws;                        // 16384
  float*    xl     = y + NH;                    // 16384
  float*    xr     = xl + NH;                   // 16384
  float*    ef     = xr + NH;                   // 540672
  float*    sc     = ef + (size_t)ETOT * H;     // 8448 (score then ex)
  float*    eamean = sc + ETOT;                 // 8
  float*    den    = eamean + EDIM;             // 256
  unsigned* mxk    = (unsigned*)(den + N_NODES);// 256
  float*    hbuf   = den + 2 * N_NODES;         // 16384
  float*    v1     = hbuf + NH;                 // 16384
  float*    v2     = v1 + NH;                   // 64

  k_init<<<(2 * NH + 2 * N_NODES + H + 255) / 256, 256, 0, stream>>>(hbuf, v1, den, mxk, v2);
  k_lin0<<<(NH + 255) / 256, 256, 0, stream>>>(x, W0, b0, y);
  k_eamean<<<1, 256, 0, stream>>>(ea, eamean);
  k_xlr<<<64, 32, 0, stream>>>(y, Wl, Wr, xl, xr);
  k_ef<<<(int)(((size_t)ETOT * H + 255) / 256), 256, 0, stream>>>(ea, eamean, We, ef);
  k_score<<<(ETOT + 255) / 256, 256, 0, stream>>>(ei, xl, xr, ef, att, sc, mxk);
  k_exden<<<(ETOT + 255) / 256, 256, 0, stream>>>(ei, mxk, sc, den);
  k_aggr<<<(int)(((size_t)ETOT * H + 255) / 256), 256, 0, stream>>>(ei, sc, den, xl, hbuf);
  k_hbias<<<(NH + 255) / 256, 256, 0, stream>>>(hbuf, bg);
  k_matvec1<<<dim3(64, 16), 256, 0, stream>>>(hbuf, W1, v1);      // dominant: 1.07 GB stream
  k_reluB<<<(NH + 255) / 256, 256, 0, stream>>>(v1, b1, NH);
  k_matvec2<<<32, 256, 0, stream>>>(v1, W2, v2);
  k_reluB<<<1, 64, 0, stream>>>(v2, b2, H);
  k_head<<<1, 32, 0, stream>>>(v2, W3, b3, out);
}
